// IntraNodeAggregator_31619549233246
// MI455X (gfx1250) — matrix operand order repaired and checked by reading, not compiled, not run on hardware
//
#include <hip/hip_runtime.h>
#include <hip/hip_bf16.h>

typedef __attribute__((ext_vector_type(16))) __bf16 v16bf;
typedef __attribute__((ext_vector_type(8)))  __bf16 v8bf;
typedef __attribute__((ext_vector_type(8)))  float  v8f;

#define BATCH   16
#define NSEQ    4096
#define DIN     256
#define DOUT    128
#define MBLK    128                 // rows per block (8 waves x 16)
#define NBLOCKS ((BATCH * NSEQ) / MBLK)   // 512
#define BLOCKS_PER_BATCH (NSEQ / MBLK)    // 32

__device__ __forceinline__ v8f wmma_bf16(v16bf a, v16bf b, v8f c) {
    return __builtin_amdgcn_wmma_f32_16x16x32_bf16(
        /*neg_a=*/false, a, /*neg_b=*/false, b,
        /*c_mod=*/(short)0, c, /*reuse_a=*/false, /*reuse_b=*/false);
}

// Convert fp32 weights (K x 128, row major) into per-lane WMMA B-fragment layout:
// fragment element f = (((ks*8 + nt)*32 + lane)*16 + i)
//   lane < 16:  n = nt*16 + lane,     k = ks*32 + {0..7, 16..23}[i]   (i = 0..15)
//   lane >= 16: n = nt*16 + lane-16,  k = ks*32 + {8..15, 24..31}[i]
__device__ __forceinline__ void fill_w_frags(const float* Wg, __bf16* lds, int KS) {
    const int total = KS * 8 * 32 * 16;
    for (int f = threadIdx.x; f < total; f += 256) {
        int i    = f & 15;
        int lane = (f >> 4) & 31;
        int nt   = (f >> 9) & 7;
        int ks   = f >> 12;
        int k = ks * 32 + ((lane >= 16) ? 8 : 0) + ((i < 8) ? i : i + 8);
        int n = nt * 16 + (lane & 15);
        lds[f] = (__bf16)Wg[k * DOUT + n];
    }
}

// A fragment (16x32 bf16) from global fp32, ISA 16-bit A layout:
// lane<16 : row=lane,   k = kb+{0..7, 16..23}
// lane>=16: row=lane-16, k = kb+{8..15, 24..31}
__device__ __forceinline__ v16bf load_a_global(const float* Xrows, int kb, int lane) {
    int hl = lane >> 4;
    const float* p = Xrows + (lane & 15) * DIN + kb + hl * 8;
    float4 a0 = *(const float4*)(p);
    float4 a1 = *(const float4*)(p + 4);
    float4 c0 = *(const float4*)(p + 16);
    float4 c1 = *(const float4*)(p + 20);
    v16bf r;
    r[0]=(__bf16)a0.x;  r[1]=(__bf16)a0.y;  r[2]=(__bf16)a0.z;  r[3]=(__bf16)a0.w;
    r[4]=(__bf16)a1.x;  r[5]=(__bf16)a1.y;  r[6]=(__bf16)a1.z;  r[7]=(__bf16)a1.w;
    r[8]=(__bf16)c0.x;  r[9]=(__bf16)c0.y;  r[10]=(__bf16)c0.z; r[11]=(__bf16)c0.w;
    r[12]=(__bf16)c1.x; r[13]=(__bf16)c1.y; r[14]=(__bf16)c1.z; r[15]=(__bf16)c1.w;
    return r;
}

// A fragment from a 16x128 bf16 LDS tile (row major).
__device__ __forceinline__ v16bf load_a_lds(const __bf16* h, int kb, int lane) {
    int hl = lane >> 4;
    const __bf16* p = h + (lane & 15) * DOUT + kb + hl * 8;
    v8bf lo = *(const v8bf*)(p);
    v8bf hi = *(const v8bf*)(p + 16);
    v16bf r;
    #pragma unroll
    for (int i = 0; i < 8; ++i) { r[i] = lo[i]; r[i + 8] = hi[i]; }
    return r;
}

__global__ void __launch_bounds__(256)
fused_mlp_kernel(const float* __restrict__ X,
                 const float* __restrict__ Wp, const float* __restrict__ bp,
                 const float* __restrict__ W1, const float* __restrict__ b1,
                 const float* __restrict__ W2, const float* __restrict__ b2,
                 float* __restrict__ Y, float* __restrict__ partials) {
    __shared__ __attribute__((aligned(32))) __bf16 s_wp[8 * 8 * 32 * 16];  // 64 KB
    __shared__ __attribute__((aligned(32))) __bf16 s_w1[4 * 8 * 32 * 16];  // 32 KB
    __shared__ __attribute__((aligned(32))) __bf16 s_w2[4 * 8 * 32 * 16];  // 32 KB
    __shared__ __attribute__((aligned(32))) __bf16 s_h[8 * 16 * DOUT];     // 32 KB
    __shared__ float s_part[8 * DOUT];                                     //  4 KB

    const int lane = threadIdx.x & 31;
    const int wave = threadIdx.x >> 5;
    const int hl   = lane >> 4;
    const int lanelo = lane & 15;

    // Stage weights into LDS in B-fragment order (bf16).
    fill_w_frags(Wp, s_wp, 8);
    fill_w_frags(W1, s_w1, 4);
    fill_w_frags(W2, s_w2, 4);
    __syncthreads();

    // Per-lane bias values for the 8 N-tiles this lane touches.
    float bp_r[8], b1_r[8], b2_r[8];
    #pragma unroll
    for (int nt = 0; nt < 8; ++nt) {
        int n = nt * 16 + lanelo;
        bp_r[nt] = bp[n]; b1_r[nt] = b1[n]; b2_r[nt] = b2[n];
    }

    const int row_base = blockIdx.x * MBLK + wave * 16;
    const float* Xrows = X + (size_t)row_base * DIN;
    __bf16* hwave = s_h + wave * 16 * DOUT;

    const v8f vzero = {0.f, 0.f, 0.f, 0.f, 0.f, 0.f, 0.f, 0.f};

    // ---- GEMM1: (16 x 256) x (256 x 128) -> H0 ----
    v8f acc[8];
    #pragma unroll
    for (int nt = 0; nt < 8; ++nt) acc[nt] = vzero;
    #pragma unroll
    for (int ks = 0; ks < 8; ++ks) {
        v16bf a = load_a_global(Xrows, ks * 32, lane);
        #pragma unroll
        for (int nt = 0; nt < 8; ++nt) {
            v16bf b = *(const v16bf*)&s_wp[((ks * 8 + nt) * 32 + lane) * 16];
            acc[nt] = wmma_bf16(a, b, acc[nt]);
        }
    }
    __syncthreads();
    // C layout -> bf16 LDS tile (transpose staging); add b_proj.
    #pragma unroll
    for (int nt = 0; nt < 8; ++nt)
        #pragma unroll
        for (int r = 0; r < 8; ++r)
            hwave[(r + hl * 8) * DOUT + nt * 16 + lanelo] = (__bf16)(acc[nt][r] + bp_r[nt]);
    __syncthreads();

    // ---- GEMM2: (16 x 128) x (128 x 128) -> ReLU -> H1 ----
    v16bf a2[4];
    #pragma unroll
    for (int ks = 0; ks < 4; ++ks) a2[ks] = load_a_lds(hwave, ks * 32, lane);
    #pragma unroll
    for (int nt = 0; nt < 8; ++nt) acc[nt] = vzero;
    #pragma unroll
    for (int ks = 0; ks < 4; ++ks)
        #pragma unroll
        for (int nt = 0; nt < 8; ++nt) {
            v16bf b = *(const v16bf*)&s_w1[((ks * 8 + nt) * 32 + lane) * 16];
            acc[nt] = wmma_bf16(a2[ks], b, acc[nt]);
        }
    __syncthreads();
    #pragma unroll
    for (int nt = 0; nt < 8; ++nt)
        #pragma unroll
        for (int r = 0; r < 8; ++r)
            hwave[(r + hl * 8) * DOUT + nt * 16 + lanelo] =
                (__bf16)fmaxf(acc[nt][r] + b1_r[nt], 0.f);
    __syncthreads();

    // ---- GEMM3: (16 x 128) x (128 x 128) -> Y ----
    #pragma unroll
    for (int ks = 0; ks < 4; ++ks) a2[ks] = load_a_lds(hwave, ks * 32, lane);
    #pragma unroll
    for (int nt = 0; nt < 8; ++nt) acc[nt] = vzero;
    #pragma unroll
    for (int ks = 0; ks < 4; ++ks)
        #pragma unroll
        for (int nt = 0; nt < 8; ++nt) {
            v16bf b = *(const v16bf*)&s_w2[((ks * 8 + nt) * 32 + lane) * 16];
            acc[nt] = wmma_bf16(a2[ks], b, acc[nt]);
        }

    // Store Y (fp32) + accumulate deterministic per-wave column sums.
    #pragma unroll
    for (int nt = 0; nt < 8; ++nt) {
        float csum = 0.f;
        #pragma unroll
        for (int r = 0; r < 8; ++r) {
            float y = acc[nt][r] + b2_r[nt];
            Y[(size_t)(row_base + r + hl * 8) * DOUT + nt * 16 + lanelo] = y;
            csum += y;
        }
        csum += __shfl_xor(csum, 16, 32);   // combine rows 0-7 with 8-15
        if (lane < 16) s_part[wave * DOUT + nt * 16 + lane] = csum;
    }
    __syncthreads();
    // Cross-wave reduce (deterministic order) -> per-block partial.
    if (threadIdx.x < DOUT) {
        float t = 0.f;
        #pragma unroll
        for (int w = 0; w < 8; ++w) t += s_part[w * DOUT + threadIdx.x];
        partials[(size_t)blockIdx.x * DOUT + threadIdx.x] = t;
    }
}

__global__ void __launch_bounds__(128)
colsum_kernel(const float* __restrict__ partials, float* __restrict__ colsum) {
    int b = blockIdx.x, c = threadIdx.x;
    float s = 0.f;
    #pragma unroll
    for (int i = 0; i < BLOCKS_PER_BATCH; ++i)
        s += partials[(size_t)(b * BLOCKS_PER_BATCH + i) * DOUT + c];
    colsum[b * DOUT + c] = s;
}

__global__ void __launch_bounds__(256)
loo_kernel(float* __restrict__ Y, const float* __restrict__ colsum) {
    // float4 granularity; per-batch float4 count = 4096*32 = 2^17
    int idx = blockIdx.x * 256 + threadIdx.x;
    int b  = idx >> 17;
    int c4 = idx & 31;
    float4 cs = ((const float4*)colsum)[b * 32 + c4];
    float4* Y4 = (float4*)Y;
    float4 y = Y4[idx];
    Y4[idx] = make_float4(cs.x - y.x, cs.y - y.y, cs.z - y.z, cs.w - y.w);
}

extern "C" void kernel_launch(void* const* d_in, const int* in_sizes, int n_in,
                              void* d_out, int out_size, void* d_ws, size_t ws_size,
                              hipStream_t stream) {
    const float* X  = (const float*)d_in[0];
    const float* Wp = (const float*)d_in[1];
    const float* bp = (const float*)d_in[2];
    const float* W1 = (const float*)d_in[3];
    const float* b1 = (const float*)d_in[4];
    const float* W2 = (const float*)d_in[5];
    const float* b2 = (const float*)d_in[6];
    float* Y = (float*)d_out;

    float* partials = (float*)d_ws;                       // 512*128 floats
    float* colsum   = partials + (size_t)NBLOCKS * DOUT;  // 16*128 floats

    fused_mlp_kernel<<<NBLOCKS, 256, 0, stream>>>(X, Wp, bp, W1, b1, W2, b2, Y, partials);
    colsum_kernel<<<BATCH, DOUT, 0, stream>>>(partials, colsum);
    int total4 = (BATCH * NSEQ * DOUT) / 4;               // 2,097,152
    loo_kernel<<<total4 / 256, 256, 0, stream>>>(Y, colsum);
}
